// ResidualCodebookCollection_48928267436256
// MI455X (gfx1250) — matrix operand order repaired and checked by reading, not compiled, not run on hardware
//
#include <hip/hip_runtime.h>
#include <hip/hip_bf16.h>

// Residual VQ for MI455X (gfx1250, wave32, WMMA bf16 path).
// B=16, D=128, L=4096, K=8, N=1024.
// d_out = z_q_aggregated (B,K,D,L) f32  ||  indices (B,L,K) i32.
//
// Main path: prepass converts codebook to bf16 + precomputes ||c||^2 into d_ws,
// so the hot loop is pure {global_load_b128(bf16 A) -> 2x v_wmma -> fused argmin}.
// Each wave holds TWO 16-token B tiles in registers (reused across all 64
// codeword chunks), halving codebook streaming per token.

#define RVQ_B 16
#define RVQ_D 128
#define RVQ_L 4096
#define RVQ_K 8
#define RVQ_N 1024
#define THREADS 64              // 2 wave32s
#define TOKENS  64              // tokens per block (32 per wave: two 16-token B tiles)
#define ST      65              // LDS token stride (odd -> conflict-free column access)

typedef __attribute__((ext_vector_type(16))) __bf16 v16bf;
typedef __attribute__((ext_vector_type(8)))  float  v8f;

union BF16x16 {
    v16bf  v;
    __bf16 h[16];
    uint4  q[2];
};

// ---------------- prepass: codebook f32 -> bf16, plus ||c||^2 ----------------
__global__ __launch_bounds__(128)
void rvq_preconv(const float* __restrict__ cb,
                 __bf16* __restrict__ cbh,
                 float*  __restrict__ gnorm)
{
    const int n = blockIdx.x * 128 + threadIdx.x;      // 0 .. K*N-1 (8192 rows)
    const float* row = cb + (size_t)n * RVQ_D;

    union { __bf16 h[RVQ_D]; uint4 q[RVQ_D / 8]; } u;
    float s = 0.f;
    #pragma unroll
    for (int j = 0; j < RVQ_D; ++j) {
        const float v = row[j];
        u.h[j] = (__bf16)v;
        s += v * v;
    }
    uint4* dst = (uint4*)(cbh + (size_t)n * RVQ_D);
    #pragma unroll
    for (int i = 0; i < RVQ_D / 8; ++i) dst[i] = u.q[i];
    gnorm[n] = s;
}

// ------------------------------- main kernel --------------------------------
template <bool PRECONV>
__global__ __launch_bounds__(THREADS)
void rvq_main(const float*  __restrict__ x,
              const float*  __restrict__ cb,
              const __bf16* __restrict__ cbh,
              const float*  __restrict__ gnorm,
              float* __restrict__ z_out,
              int*   __restrict__ idx_out)
{
    __shared__ float s_xres [RVQ_D][ST];   // residual, transposed [d][token]
    __shared__ float s_cnorm[RVQ_N];
    __shared__ int   s_idx  [TOKENS];

    const int tid  = threadIdx.x;
    const int lane = tid & 31;
    const int wv   = tid >> 5;
    const int t0   = wv * 32;                    // wave owns tokens t0 .. t0+31

    const int blk  = blockIdx.x;                 // 16 batches * 64 tiles = 1024 blocks
    const int b    = blk >> 6;
    const int l0   = (blk & 63) * TOKENS;

    // ---- load x tile into LDS residual (coalesced along L) ----
    const float* xb = x + ((size_t)b * RVQ_D) * RVQ_L + l0;
    #pragma unroll 4
    for (int e = tid; e < RVQ_D * TOKENS; e += THREADS) {
        const int d = e >> 6;
        const int m = e & 63;
        s_xres[d][m] = xb[(size_t)d * RVQ_L + m];
    }
    __syncthreads();

    const int arow = lane & 15;                  // A row = codeword within 16-tile
    const int koff = (lane >= 16) ? 8 : 0;       // 16-bit A-layout K sub-offset

    for (int k = 0; k < RVQ_K; ++k) {
        const float*  cbk  = cb  + (size_t)k * RVQ_N * RVQ_D;
        const __bf16* cbhk = cbh + (size_t)k * RVQ_N * RVQ_D;

        // ---- codeword norms into LDS ----
        if (PRECONV) {
            for (int n = tid; n < RVQ_N; n += THREADS)
                s_cnorm[n] = gnorm[k * RVQ_N + n];
        } else {
            for (int n = tid; n < RVQ_N; n += THREADS) {
                const float4* row = (const float4*)(cbk + (size_t)n * RVQ_D);
                float s = 0.f;
                #pragma unroll 8
                for (int j = 0; j < RVQ_D / 4; ++j) {
                    const float4 v = row[j];
                    s += v.x * v.x + v.y * v.y + v.z * v.z + v.w * v.w;
                }
                s_cnorm[n] = s;
            }
        }
        __syncthreads();

        // ---- B fragments: two 16-token tiles per wave, reused for all chunks ----
        // lane%16 = token of the tile, element i = K (table order: 8h+i, 16+8h+(i-8)).
        v16bf bfrag[2][4];
        #pragma unroll
        for (int t = 0; t < 2; ++t) {
            const int tb = t0 + t * 16;
            #pragma unroll
            for (int c = 0; c < 4; ++c) {
                const int d = 32 * c + koff;
                BF16x16 u;
                #pragma unroll
                for (int i = 0; i < 16; ++i)
                    u.h[i] = (__bf16)s_xres[d + i + (i >= 8 ? 8 : 0)][tb + arow];
                bfrag[t][c] = u.v;
            }
        }

        // ---- scan all 1024 codewords ----
        float bestS0 = 3.402823466e+38f, bestS1 = 3.402823466e+38f;
        int   bestI0 = 0, bestI1 = 0;

        for (int nc = 0; nc < 64; ++nc) {
            v8f acc0 = {}, acc1 = {};
            #pragma unroll
            for (int c = 0; c < 4; ++c) {
                const int kb = 32 * c + koff;
                BF16x16 a;
                if (PRECONV) {
                    const __bf16* rowh = cbhk + (size_t)(nc * 16 + arow) * RVQ_D;
                    if (c == 0 && nc + 1 < 64)
                        __builtin_prefetch(cbhk + (size_t)((nc + 1) * 16 + arow) * RVQ_D, 0, 3);
                    a.q[0] = *(const uint4*)(rowh + kb);
                    a.q[1] = *(const uint4*)(rowh + kb + 16);
                } else {
                    const float* rowp = cbk + (size_t)(nc * 16 + arow) * RVQ_D;
                    if (c == 0 && nc + 1 < 64)
                        __builtin_prefetch(cbk + (size_t)((nc + 1) * 16 + arow) * RVQ_D, 0, 3);
                    #pragma unroll
                    for (int i = 0; i < 8; ++i) a.h[i]     = (__bf16)rowp[kb + i];
                    #pragma unroll
                    for (int i = 0; i < 8; ++i) a.h[8 + i] = (__bf16)rowp[kb + 16 + i];
                }
                acc0 = __builtin_amdgcn_wmma_f32_16x16x32_bf16(
                           false, a.v, false, bfrag[0][c], (short)0, acc0, false, false);
                acc1 = __builtin_amdgcn_wmma_f32_16x16x32_bf16(
                           false, a.v, false, bfrag[1][c], (short)0, acc1, false, false);
            }
            // D tile row r -> codeword nc*16 + r (+8 for upper half-wave)
            const int nb = nc * 16 + koff;
            #pragma unroll
            for (int r = 0; r < 8; ++r) {
                const int n = nb + r;
                const float cn = s_cnorm[n];
                const float sa = cn - 2.0f * acc0[r];
                const float sb = cn - 2.0f * acc1[r];
                if (sa < bestS0) { bestS0 = sa; bestI0 = n; }  // strict <: lowest idx on tie
                if (sb < bestS1) { bestS1 = sb; bestI1 = n; }
            }
        }

        // ---- combine half-wave candidates (same token lives in lane, lane^16) ----
        {
            const float oS = __shfl_xor(bestS0, 16, 32);
            const int   oI = __shfl_xor(bestI0, 16, 32);
            if (oS < bestS0 || (oS == bestS0 && oI < bestI0)) { bestS0 = oS; bestI0 = oI; }
        }
        {
            const float oS = __shfl_xor(bestS1, 16, 32);
            const int   oI = __shfl_xor(bestI1, 16, 32);
            if (oS < bestS1 || (oS == bestS1 && oI < bestI1)) { bestS1 = oS; bestI1 = oI; }
        }
        if (lane < 16) {
            const int ta = t0 + lane;
            const int tbk = t0 + 16 + lane;
            s_idx[ta]  = bestI0;
            s_idx[tbk] = bestI1;
            idx_out[((size_t)b * RVQ_L + (l0 + ta))  * RVQ_K + k] = bestI0;
            idx_out[((size_t)b * RVQ_L + (l0 + tbk)) * RVQ_K + k] = bestI1;
        }
        __syncthreads();

        // ---- residual update: subtract chosen f32 codeword rows (coalesced reads) ----
        for (int m = 0; m < TOKENS; ++m) {
            const float* crow = cbk + (size_t)s_idx[m] * RVQ_D;
            for (int d = tid; d < RVQ_D; d += THREADS)
                s_xres[d][m] -= crow[d];
        }
        __syncthreads();

        // ---- z_cum = x - residual (x tile is L2-hot), coalesced along L ----
        float* zk = z_out + (((size_t)b * RVQ_K + k) * RVQ_D) * RVQ_L + l0;
        #pragma unroll 4
        for (int e = tid; e < RVQ_D * TOKENS; e += THREADS) {
            const int d = e >> 6;
            const int m = e & 63;
            zk[(size_t)d * RVQ_L + m] = xb[(size_t)d * RVQ_L + m] - s_xres[d][m];
        }
        __syncthreads();
    }
}

extern "C" void kernel_launch(void* const* d_in, const int* in_sizes, int n_in,
                              void* d_out, int out_size, void* d_ws, size_t ws_size,
                              hipStream_t stream) {
    (void)in_sizes; (void)n_in; (void)out_size;

    const float* x  = (const float*)d_in[0];   // (B, D, L) f32
    const float* cb = (const float*)d_in[1];   // (K, N, D) f32

    float* z_out   = (float*)d_out;                                                  // (B,K,D,L)
    int*   idx_out = (int*)((float*)d_out + (size_t)RVQ_B * RVQ_K * RVQ_D * RVQ_L);  // (B,L,K)

    const int grid = RVQ_B * (RVQ_L / TOKENS);         // 1024 blocks x 64 threads

    const size_t ws_bf16  = (size_t)RVQ_K * RVQ_N * RVQ_D * sizeof(__bf16);  // 2 MB
    const size_t ws_norm  = (size_t)RVQ_K * RVQ_N * sizeof(float);           // 32 KB
    const size_t ws_need  = ws_bf16 + ws_norm;

    if (ws_size >= ws_need) {
        __bf16* cbh   = (__bf16*)d_ws;
        float*  gnorm = (float*)((char*)d_ws + ws_bf16);
        rvq_preconv<<<(RVQ_K * RVQ_N) / 128, 128, 0, stream>>>(cb, cbh, gnorm);
        rvq_main<true><<<grid, THREADS, 0, stream>>>(x, cb, cbh, gnorm, z_out, idx_out);
    } else {
        rvq_main<false><<<grid, THREADS, 0, stream>>>(x, cb, nullptr, nullptr, z_out, idx_out);
    }
}
